// GraphTransformerAccidentModel_1168231105210
// MI455X (gfx1250) — matrix operand order repaired and checked by reading, not compiled, not run on hardware
//
#include <hip/hip_runtime.h>
#include <hip/hip_bf16.h>
#include <math.h>

typedef __attribute__((ext_vector_type(16))) __bf16 v16bf;
typedef __attribute__((ext_vector_type(8)))  __bf16 v8bf;
typedef __attribute__((ext_vector_type(4)))  __bf16 v4bf;
typedef __attribute__((ext_vector_type(8)))  float  v8f;
typedef __attribute__((ext_vector_type(4)))  unsigned int u32x4;
typedef __attribute__((ext_vector_type(8)))  int i32x8;
typedef __attribute__((ext_vector_type(4)))  int i32x4;

#define T_FRAMES 256
#define N_OBJ    30
#define M_ROWS   (T_FRAMES * N_OBJ)   // 7680
#define K_FEAT   4096
#define DH       256
#define N_OUT    1024                 // q|k|v|s concatenated
#define DFF      1024

// Use the Tensor Data Mover for GEMM tile staging when the builtin exists.
#if defined(__has_builtin)
#if __has_builtin(__builtin_amdgcn_tensor_load_to_lds)
#define USE_TDM 1
#endif
#endif
#ifndef USE_TDM
#define USE_TDM 0
#endif

// ---------------- precision conversion kernels ----------------

__global__ void cvt_x(const float* __restrict__ src, __bf16* __restrict__ dst, int n) {
    int i = blockIdx.x * blockDim.x + threadIdx.x;
    if (i < n) dst[i] = (__bf16)src[i];
}

// Build W^T (1024 x 4096) bf16 from the four (4097 x 256) fp32 matrices,
// plus fp32 depth-row (K=4096) weights and biases for the GEMM epilogue.
__global__ void cvt_w(const float* __restrict__ Wq, const float* __restrict__ Wk,
                      const float* __restrict__ Wv, const float* __restrict__ Ws,
                      const float* __restrict__ bq, const float* __restrict__ bk,
                      const float* __restrict__ bv, const float* __restrict__ bs,
                      __bf16* __restrict__ WbT, float* __restrict__ wlast,
                      float* __restrict__ bias) {
    int i = blockIdx.x * blockDim.x + threadIdx.x;  // over 1024*4096
    if (i >= N_OUT * K_FEAT) return;
    int n = i >> 12;          // output column 0..1023
    int k = i & 4095;         // K index 0..4095
    int mat = n >> 8;
    int h = n & 255;
    const float* W = (mat == 0) ? Wq : (mat == 1) ? Wk : (mat == 2) ? Wv : Ws;
    WbT[(size_t)n * K_FEAT + k] = (__bf16)W[(size_t)k * DH + h];
    if (k == 0) {
        wlast[n] = W[(size_t)K_FEAT * DH + h];
        const float* b = (mat == 0) ? bq : (mat == 1) ? bk : (mat == 2) ? bv : bs;
        bias[n] = b[h];
    }
}

// ---------------- main bf16 WMMA GEMM: (7680x4096) x (4096x1024) ----------------

#define BM 128
#define BN 128
#define BK 32
#define LSTR 40   // bf16 elems per LDS row (80 B = 64 B data + 16 B pad)

#if USE_TDM
// Issue one TDM descriptor moving a 128-row x 32-col bf16 tile (row stride
// `stride_elems` in memory) into LDS at byte offset `lds_off`, padding the
// destination by 4 DWORDs after every 16 DWORDs so LDS rows are 80 B.
// D# layout per CDNA5 ISA.
__device__ __forceinline__ void tdm_load_tile_2d(
    unsigned int lds_off, const __bf16* gptr,
    unsigned int tensor_d0, unsigned int tensor_d1,
    unsigned long long stride_elems)
{
    unsigned long long ga = (unsigned long long)gptr;
    u32x4 g0;
    g0[0] = 1u;                                            // count=1 (valid user D#)
    g0[1] = lds_off;                                       // lds_addr (bytes)
    g0[2] = (unsigned int)(ga & 0xFFFFFFFFu);              // global_addr[31:0]
    g0[3] = (unsigned int)((ga >> 32) & 0x1FFFFFFu)        // global_addr[56:32]
          | (2u << 30);                                    // type=2 ("image")
    i32x8 g1;
    g1[0] = (int)((1u << 16)        // data_size = 1 -> 2 bytes/elem
                | (1u << 20)        // pad_enable
                | (3u << 22)        // pad_interval: every 16 DWORDs (64 B)
                | (3u << 25));      // pad_amount: 4 DWORDs (16 B)
    g1[1] = (int)((tensor_d0 & 0xFFFFu) << 16);            // tensor_dim0[15:0]
    g1[2] = (int)((tensor_d0 >> 16) | ((tensor_d1 & 0xFFFFu) << 16));
    g1[3] = (int)((tensor_d1 >> 16) | (32u << 16));        // tile_dim0 = 32 (K)
    g1[4] = (int)128u;                                     // tile_dim1 = 128 rows
    g1[5] = (int)(unsigned int)(stride_elems & 0xFFFFFFFFull);   // dim0_stride lo
    g1[6] = (int)(unsigned int)((stride_elems >> 32) & 0xFFFFull);
    g1[7] = 0;
    i32x4 z4 = {0, 0, 0, 0};
#if __clang_major__ >= 23
    i32x8 z8 = {0, 0, 0, 0, 0, 0, 0, 0};
    __builtin_amdgcn_tensor_load_to_lds(g0, g1, z4, z4, z8, 0);
#else
    __builtin_amdgcn_tensor_load_to_lds(g0, g1, z4, z4, 0);
#endif
}
#endif // USE_TDM

__global__ __launch_bounds__(256) void gemm_qkvs_wmma(
    const __bf16* __restrict__ Xb,    // (7680, 4096)
    const __bf16* __restrict__ WbT,   // (1024, 4096)  (pre-transposed weights)
    const float*  __restrict__ wlast, // (1024) depth-row weights
    const float*  __restrict__ bias,  // (1024)
    const float*  __restrict__ depth, // (7680)
    float*        __restrict__ C)     // (7680, 1024)
{
    __shared__ __attribute__((aligned(16))) __bf16 smA[2][BM * LSTR];
    __shared__ __attribute__((aligned(16))) __bf16 smB[2][BN * LSTR];

    const int tid  = threadIdx.x;
    const int lane = tid & 31;
    const int wave = tid >> 5;       // 0..7
    const int wr   = wave >> 2;      // wave row (M): 0..1, 64 rows each
    const int wc   = wave & 3;       // wave col (N): 0..3, 32 cols each
    const int m0   = blockIdx.x * BM;
    const int n0   = blockIdx.y * BN;

    // WMMA fragment addressing (16-bit A/B layouts per CDNA5 ISA 7.12.2)
    const int mlane = lane & 15;
    const int hi    = lane >> 4;        // 0: lanes 0-15, 1: lanes 16-31
    const int kbA   = hi ? 8 : 0;       // A: K chunks {kbA..kbA+7, kbA+16..kbA+23}
    const int kbB   = hi ? 8 : 0;       // B: K chunks {kbB..kbB+7, kbB+16..kbB+23}

    v8f acc[4][2];
    for (int i = 0; i < 4; ++i)
        for (int j = 0; j < 2; ++j)
            for (int r = 0; r < 8; ++r) acc[i][j][r] = 0.0f;

    const int NK = K_FEAT / BK;  // 128

#if USE_TDM
    const bool loader = (wave == 0);
    const unsigned int ldsA[2] = { (unsigned int)(unsigned long long)&smA[0][0],
                                   (unsigned int)(unsigned long long)&smA[1][0] };
    const unsigned int ldsB[2] = { (unsigned int)(unsigned long long)&smB[0][0],
                                   (unsigned int)(unsigned long long)&smB[1][0] };
    const __bf16* gA = Xb  + (size_t)m0 * K_FEAT;  // tile row base
    const __bf16* gB = WbT + (size_t)n0 * K_FEAT;

    if (loader) {   // DMA tile 0 into buffer 0 via the Tensor Data Mover
        tdm_load_tile_2d(ldsA[0], gA, K_FEAT, M_ROWS, K_FEAT);
        tdm_load_tile_2d(ldsB[0], gB, K_FEAT, N_OUT,  K_FEAT);
        __builtin_amdgcn_s_wait_tensorcnt(0);
    }
    __syncthreads();
#else
    // manual staging: each thread moves 4 x 8B chunks per tile
    const int lrow = tid >> 3;          // 0..31 (row within 32-row group)
    const int lcol = (tid & 7) * 4;     // bf16 elem offset within 32-elem row
    {
        for (int q = 0; q < 4; ++q) {
            int r = lrow + q * 32;
            v4bf a = *(const v4bf*)(Xb  + (size_t)(m0 + r) * K_FEAT + lcol);
            v4bf b = *(const v4bf*)(WbT + (size_t)(n0 + r) * K_FEAT + lcol);
            *(v4bf*)(&smA[0][r * LSTR + lcol]) = a;
            *(v4bf*)(&smB[0][r * LSTR + lcol]) = b;
        }
    }
    __syncthreads();
    v4bf ra[4], rb[4];
#endif

    for (int kt = 0; kt < NK; ++kt) {
        const int cur = kt & 1;
        const int nxt = cur ^ 1;

#if USE_TDM
        if (kt + 1 < NK && loader) {
            const int k0 = (kt + 1) * BK;
            tdm_load_tile_2d(ldsA[nxt], gA + k0, K_FEAT, M_ROWS, K_FEAT);
            tdm_load_tile_2d(ldsB[nxt], gB + k0, K_FEAT, N_OUT,  K_FEAT);
        }
#else
        if (kt + 1 < NK) {
            const int k0 = (kt + 1) * BK;
            for (int q = 0; q < 4; ++q) {
                int r = lrow + q * 32;
                ra[q] = *(const v4bf*)(Xb  + (size_t)(m0 + r) * K_FEAT + k0 + lcol);
                rb[q] = *(const v4bf*)(WbT + (size_t)(n0 + r) * K_FEAT + k0 + lcol);
            }
            if (kt + 2 < NK) {  // lowers to global_prefetch_b8 on gfx1250
                __builtin_prefetch(Xb  + (size_t)(m0 + lrow) * K_FEAT + (size_t)(kt + 2) * BK, 0, 1);
                __builtin_prefetch(WbT + (size_t)(n0 + lrow) * K_FEAT + (size_t)(kt + 2) * BK, 0, 1);
            }
        }
#endif

        // compute on current buffer
        v16bf af[4], bfv[2];
        for (int i = 0; i < 4; ++i) {
            const __bf16* p = &smA[cur][(wr * 64 + i * 16 + mlane) * LSTR + kbA];
            *((v8bf*)&af[i])     = *(const v8bf*)p;          // K = kbA .. kbA+7
            *((v8bf*)&af[i] + 1) = *(const v8bf*)(p + 16);   // K = kbA+16 .. kbA+23
        }
        for (int j = 0; j < 2; ++j) {
            const __bf16* p = &smB[cur][(wc * 32 + j * 16 + mlane) * LSTR + kbB];
            *((v8bf*)&bfv[j])     = *(const v8bf*)p;         // K = kbB .. kbB+7
            *((v8bf*)&bfv[j] + 1) = *(const v8bf*)(p + 16);  // K = kbB+16 .. kbB+23
        }
        for (int i = 0; i < 4; ++i)
            for (int j = 0; j < 2; ++j)
                acc[i][j] = __builtin_amdgcn_wmma_f32_16x16x32_bf16(
                    false, af[i], false, bfv[j], (short)0, acc[i][j], false, false);

        if (kt + 1 < NK) {
#if USE_TDM
            if (loader) __builtin_amdgcn_s_wait_tensorcnt(0);
            __syncthreads();
#else
            for (int q = 0; q < 4; ++q) {
                int r = lrow + q * 32;
                *(v4bf*)(&smA[nxt][r * LSTR + lcol]) = ra[q];
                *(v4bf*)(&smB[nxt][r * LSTR + lcol]) = rb[q];
            }
            __syncthreads();
#endif
        }
    }

    // epilogue: + depth*W[4096,:] + bias (fp32), write fp32
    for (int i = 0; i < 4; ++i) {
        for (int r = 0; r < 8; ++r) {
            int row = m0 + wr * 64 + i * 16 + hi * 8 + r;
            float d = depth[row];
            for (int j = 0; j < 2; ++j) {
                int col = n0 + wc * 32 + j * 16 + mlane;
                C[(size_t)row * N_OUT + col] = acc[i][j][r] + d * wlast[col] + bias[col];
            }
        }
    }
}

// ---------------- per-frame spatial attention + mean pool (fp32) ----------------

__global__ __launch_bounds__(256) void spatial_attn(
    const float* __restrict__ QKVS, float* __restrict__ pooled)
{
    __shared__ float sk[N_OBJ][257];   // stride 257 -> conflict-free columns
    __shared__ float sv[N_OBJ][257];
    __shared__ float attn[N_OBJ][N_OBJ];
    const int t   = blockIdx.x;
    const int tid = threadIdx.x;
    const float* base = QKVS + (size_t)t * N_OBJ * N_OUT;

    for (int e = tid; e < N_OBJ * DH; e += 256) {
        int n = e >> 8, h = e & 255;
        sk[n][h] = base[n * N_OUT + 256 + h];
        sv[n][h] = base[n * N_OUT + 512 + h];
    }
    __syncthreads();

    for (int p = tid; p < N_OBJ * N_OBJ; p += 256) {
        int n = p / N_OBJ, m = p % N_OBJ;
        float s;
        if (n == m) s = -1e30f;              // self-edge masked
        else {
            const float* qn = base + n * N_OUT;   // q row (L2-resident)
            float a = 0.f;
            for (int d = 0; d < DH; ++d) a += qn[d] * sk[m][d];
            s = a * (1.0f / 16.0f);               // 1/sqrt(256)
        }
        attn[n][m] = s;
    }
    __syncthreads();

    if (tid < N_OBJ) {
        float mx = -1e30f;
        for (int m = 0; m < N_OBJ; ++m) mx = fmaxf(mx, attn[tid][m]);
        float sum = 0.f;
        for (int m = 0; m < N_OBJ; ++m) { float e = __expf(attn[tid][m] - mx); attn[tid][m] = e; sum += e; }
        float inv = 1.0f / sum;
        for (int m = 0; m < N_OBJ; ++m) attn[tid][m] *= inv;
    }
    __syncthreads();

    // thread h: msg + skip + ReLU + mean-pool over n
    {
        int h = tid;   // blockDim == DH == 256
        float pool = 0.f;
        for (int n = 0; n < N_OBJ; ++n) {
            float msg = 0.f;
            for (int m = 0; m < N_OBJ; ++m) msg += attn[n][m] * sv[m][h];
            float val = base[n * N_OUT + 768 + h] + msg;   // x@Ws+bs already in col block 3
            pool += fmaxf(val, 0.f);
        }
        pooled[t * DH + h] = pool * (1.0f / N_OBJ);
    }
}

// ---------------- temporal transformer block (fp32, tiny) ----------------

__global__ __launch_bounds__(256) void t_qkv(
    const float* __restrict__ P, const float* __restrict__ tWq,
    const float* __restrict__ tWk, const float* __restrict__ tWv,
    float* __restrict__ q2, float* __restrict__ k2, float* __restrict__ v2)
{
    __shared__ float sx[DH];
    int i = blockIdx.x, h = threadIdx.x;
    sx[h] = P[i * DH + h];
    __syncthreads();
    float aq = 0.f, ak = 0.f, av = 0.f;
    for (int d = 0; d < DH; ++d) {
        float x = sx[d];
        aq += x * tWq[d * DH + h];
        ak += x * tWk[d * DH + h];
        av += x * tWv[d * DH + h];
    }
    q2[i * DH + h] = aq; k2[i * DH + h] = ak; v2[i * DH + h] = av;
}

__global__ __launch_bounds__(256) void t_attn(
    const float* __restrict__ q2, const float* __restrict__ k2,
    const float* __restrict__ v2, float* __restrict__ a1)
{
    __shared__ float sq[DH];
    __shared__ float sp[T_FRAMES];
    __shared__ float red[256];
    int i = blockIdx.x, tid = threadIdx.x;
    sq[tid] = q2[i * DH + tid];
    __syncthreads();
    float a = 0.f;
    for (int d = 0; d < DH; ++d) a += sq[d] * k2[tid * DH + d];
    float s = a * (1.0f / 16.0f);
    red[tid] = s; __syncthreads();
    for (int off = 128; off > 0; off >>= 1) { if (tid < off) red[tid] = fmaxf(red[tid], red[tid + off]); __syncthreads(); }
    float mx = red[0]; __syncthreads();
    float e = __expf(s - mx);
    red[tid] = e; __syncthreads();
    for (int off = 128; off > 0; off >>= 1) { if (tid < off) red[tid] += red[tid + off]; __syncthreads(); }
    float inv = 1.0f / red[0]; __syncthreads();
    sp[tid] = e * inv;
    __syncthreads();
    float o = 0.f;
    for (int j = 0; j < T_FRAMES; ++j) o += sp[j] * v2[(size_t)j * DH + tid];
    a1[i * DH + tid] = o;
}

__global__ __launch_bounds__(256) void t_y(
    const float* __restrict__ a1, const float* __restrict__ tWo,
    const float* __restrict__ P, const float* __restrict__ g,
    const float* __restrict__ b, float* __restrict__ Y)
{
    __shared__ float sa[DH];
    __shared__ float red[256];
    int i = blockIdx.x, h = threadIdx.x;
    sa[h] = a1[i * DH + h];
    __syncthreads();
    float acc = 0.f;
    for (int d = 0; d < DH; ++d) acc += sa[d] * tWo[d * DH + h];
    float r = P[i * DH + h] + acc;
    red[h] = r; __syncthreads();
    for (int off = 128; off > 0; off >>= 1) { if (h < off) red[h] += red[h + off]; __syncthreads(); }
    float mu = red[0] * (1.0f / DH); __syncthreads();
    float dr = r - mu;
    red[h] = dr * dr; __syncthreads();
    for (int off = 128; off > 0; off >>= 1) { if (h < off) red[h] += red[h + off]; __syncthreads(); }
    float var = red[0] * (1.0f / DH);
    Y[i * DH + h] = dr * rsqrtf(var + 1e-5f) * g[h] + b[h];
}

__global__ __launch_bounds__(256) void t_ffn1(
    const float* __restrict__ Y, const float* __restrict__ fW1,
    const float* __restrict__ fb1, float* __restrict__ H)
{
    __shared__ float sy[DH];
    int i = blockIdx.x, tid = threadIdx.x;
    sy[tid] = Y[i * DH + tid];
    __syncthreads();
    float a0 = 0.f, a1v = 0.f, a2 = 0.f, a3 = 0.f;
    for (int d = 0; d < DH; ++d) {
        float x = sy[d];
        const float* w = fW1 + (size_t)d * DFF + tid;
        a0 += x * w[0]; a1v += x * w[256]; a2 += x * w[512]; a3 += x * w[768];
    }
    float* o = H + (size_t)i * DFF + tid;
    o[0]   = fmaxf(a0  + fb1[tid],       0.f);
    o[256] = fmaxf(a1v + fb1[tid + 256], 0.f);
    o[512] = fmaxf(a2  + fb1[tid + 512], 0.f);
    o[768] = fmaxf(a3  + fb1[tid + 768], 0.f);
}

__global__ __launch_bounds__(256) void t_out(
    const float* __restrict__ H, const float* __restrict__ fW2,
    const float* __restrict__ fb2, const float* __restrict__ Y,
    const float* __restrict__ g, const float* __restrict__ b,
    const float* __restrict__ cw, const float* __restrict__ cb,
    float* __restrict__ out)
{
    __shared__ float sh[DFF];
    __shared__ float red[256];
    int i = blockIdx.x, h = threadIdx.x;
    for (int f = h; f < DFF; f += 256) sh[f] = H[(size_t)i * DFF + f];
    __syncthreads();
    float acc = 0.f;
    for (int f = 0; f < DFF; ++f) acc += sh[f] * fW2[(size_t)f * DH + h];
    float r = Y[i * DH + h] + acc + fb2[h];
    red[h] = r; __syncthreads();
    for (int off = 128; off > 0; off >>= 1) { if (h < off) red[h] += red[h + off]; __syncthreads(); }
    float mu = red[0] * (1.0f / DH); __syncthreads();
    float dr = r - mu;
    red[h] = dr * dr; __syncthreads();
    for (int off = 128; off > 0; off >>= 1) { if (h < off) red[h] += red[h + off]; __syncthreads(); }
    float var = red[0] * (1.0f / DH); __syncthreads();
    float z = dr * rsqrtf(var + 1e-5f) * g[h] + b[h];
    red[h] = z * cw[h]; __syncthreads();
    for (int off = 128; off > 0; off >>= 1) { if (h < off) red[h] += red[h + off]; __syncthreads(); }
    if (h == 0) {
        float logit = red[0] + cb[0];
        out[i] = 1.0f / (1.0f + __expf(-logit));
        out[T_FRAMES + i] = 0.0f;    // uncertainty is exactly 0 in eval mode
    }
}

// ---------------- host-side orchestration ----------------

static constexpr size_t alup(size_t x) { return (x + 255) & ~(size_t)255; }

extern "C" void kernel_launch(void* const* d_in, const int* in_sizes, int n_in,
                              void* d_out, int out_size, void* d_ws, size_t ws_size,
                              hipStream_t stream) {
    const float* feat  = (const float*)d_in[0];
    const float* depth = (const float*)d_in[1];
    const float* Wq = (const float*)d_in[2];  const float* bq = (const float*)d_in[3];
    const float* Wk = (const float*)d_in[4];  const float* bk = (const float*)d_in[5];
    const float* Wv = (const float*)d_in[6];  const float* bv = (const float*)d_in[7];
    const float* Ws = (const float*)d_in[8];  const float* bs = (const float*)d_in[9];
    const float* tWq = (const float*)d_in[10];
    const float* tWk = (const float*)d_in[11];
    const float* tWv = (const float*)d_in[12];
    const float* tWo = (const float*)d_in[13];
    const float* ln1g = (const float*)d_in[14]; const float* ln1b = (const float*)d_in[15];
    const float* ln2g = (const float*)d_in[16]; const float* ln2b = (const float*)d_in[17];
    const float* fW1 = (const float*)d_in[18];  const float* fb1 = (const float*)d_in[19];
    const float* fW2 = (const float*)d_in[20];  const float* fb2 = (const float*)d_in[21];
    const float* cw  = (const float*)d_in[22];  const float* cb  = (const float*)d_in[23];
    float* out = (float*)d_out;

    char* ws = (char*)d_ws;
    size_t off = 0;
    __bf16* XB   = (__bf16*)(ws + off); off = alup(off + (size_t)M_ROWS * K_FEAT * 2);
    __bf16* WBT  = (__bf16*)(ws + off); off = alup(off + (size_t)N_OUT * K_FEAT * 2);
    float* WLAST = (float*)(ws + off);  off = alup(off + (size_t)N_OUT * 4);
    float* BIAS  = (float*)(ws + off);  off = alup(off + (size_t)N_OUT * 4);
    float* QKVS  = (float*)(ws + off);  off = alup(off + (size_t)M_ROWS * N_OUT * 4);
    float* POOL  = (float*)(ws + off);  off = alup(off + (size_t)T_FRAMES * DH * 4);
    float* Q2    = (float*)(ws + off);  off = alup(off + (size_t)T_FRAMES * DH * 4);
    float* K2    = (float*)(ws + off);  off = alup(off + (size_t)T_FRAMES * DH * 4);
    float* V2    = (float*)(ws + off);  off = alup(off + (size_t)T_FRAMES * DH * 4);
    float* A1    = (float*)(ws + off);  off = alup(off + (size_t)T_FRAMES * DH * 4);
    float* Y     = (float*)(ws + off);  off = alup(off + (size_t)T_FRAMES * DH * 4);
    float* Hb    = (float*)(ws + off);  off = alup(off + (size_t)T_FRAMES * DFF * 4);

    const int nx = M_ROWS * K_FEAT;
    cvt_x<<<(nx + 255) / 256, 256, 0, stream>>>(feat, XB, nx);
    const int nw = N_OUT * K_FEAT;
    cvt_w<<<(nw + 255) / 256, 256, 0, stream>>>(Wq, Wk, Wv, Ws, bq, bk, bv, bs,
                                                WBT, WLAST, BIAS);
    gemm_qkvs_wmma<<<dim3(M_ROWS / BM, N_OUT / BN), 256, 0, stream>>>(
        XB, WBT, WLAST, BIAS, depth, QKVS);
    spatial_attn<<<T_FRAMES, 256, 0, stream>>>(QKVS, POOL);
    t_qkv<<<T_FRAMES, 256, 0, stream>>>(POOL, tWq, tWk, tWv, Q2, K2, V2);
    t_attn<<<T_FRAMES, 256, 0, stream>>>(Q2, K2, V2, A1);
    t_y<<<T_FRAMES, 256, 0, stream>>>(A1, tWo, POOL, ln1g, ln1b, Y);
    t_ffn1<<<T_FRAMES, 256, 0, stream>>>(Y, fW1, fb1, Hb);
    t_out<<<T_FRAMES, 256, 0, stream>>>(Hb, fW2, fb2, Y, ln2g, ln2b, cw, cb, out);
}
